// ParameterGenerator_11536282157778
// MI455X (gfx1250) — matrix operand order repaired and checked by reading, not compiled, not run on hardware
//
#include <hip/hip_runtime.h>

#define N_DEC   24
#define IN_DIM  768
#define HID     128
#define OUT_DIM 99136
#define BATCH   64
#define OTILES  (OUT_DIM / 16)   // 6196 exactly

typedef __attribute__((ext_vector_type(16))) __bf16 v16bf;
typedef __attribute__((ext_vector_type(8)))  __bf16 v8bf;
typedef __attribute__((ext_vector_type(8)))  float  v8f;

// Split f32 into bf16 hi (exact top-16-bit truncation) + bf16 lo (RNE of residual).
// f = hi + lo with ~2^-17 relative error.
__device__ __forceinline__ void split_bf16(float f, __bf16& hi, __bf16& lo) {
  unsigned hu = __float_as_uint(f) & 0xFFFF0000u;
  float r = f - __uint_as_float(hu);          // exact
  hi = __builtin_bit_cast(__bf16, (unsigned short)(hu >> 16));
  lo = (__bf16)r;
}

// ---------------- Kernel 1: h = relu(x @ w1 + b1), stored pre-split bf16 hi/lo ---
// grid (BATCH, N_DEC), block 128 (one thread per HID column).
__global__ __launch_bounds__(128) void gen_hidden_kernel(
    const float* __restrict__ x, const float* __restrict__ w1,
    const float* __restrict__ b1,
    __bf16* __restrict__ h_hi, __bf16* __restrict__ h_lo) {
  const int m = blockIdx.x;
  const int n = blockIdx.y;
  const int t = threadIdx.x;                       // 0..127
  const float* xr = x + m * IN_DIM;                // block-uniform -> scalar loads
  const float* w  = w1 + (size_t)n * IN_DIM * HID + t;
  float acc = b1[n * HID + t];
#pragma unroll 8
  for (int k = 0; k < IN_DIM; ++k)
    acc = fmaf(xr[k], w[(size_t)k * HID], acc);
  acc = fmaxf(acc, 0.0f);
  __bf16 hi, lo;
  split_bf16(acc, hi, lo);
  const int idx = (n * BATCH + m) * HID + t;
  h_hi[idx] = hi;
  h_lo[idx] = lo;
}

// ---------------- Kernel 2: out = h @ w2 + b2 via bf16x3 WMMA ---------------------
// grid (ceil(OTILES/16), N_DEC), block 128 = 4 waves; each wave: 4 o-tiles of 64x16.
__global__ __launch_bounds__(128, 1) void gen_out_kernel(
    const __bf16* __restrict__ h_hi, const __bf16* __restrict__ h_lo,
    const float* __restrict__ w2, const float* __restrict__ b2,
    float* __restrict__ out) {
  const int n     = blockIdx.y;
  const int lane  = threadIdx.x & 31;
  const int wave  = threadIdx.x >> 5;
  const int laneN = lane & 15;    // column within tile / A row
  const int laneH = lane >> 4;    // half-wave select

  // ---- A operands (all of h[n], hi+lo) resident in registers: 4 m-tiles x 4 k-tiles
  v16bf Ahi[4][4], Alo[4][4];
#pragma unroll
  for (int mt = 0; mt < 4; ++mt) {
    const int m = mt * 16 + laneN;
    const __bf16* bh = h_hi + (size_t)(n * BATCH + m) * HID;
    const __bf16* bl = h_lo + (size_t)(n * BATCH + m) * HID;
#pragma unroll
    for (int kt = 0; kt < 4; ++kt) {
      const int k0 = kt * 32 + laneH * 16;   // lane<16: K 0..15 ; lane>=16: 16..31
      v8bf h0 = *(const v8bf*)(bh + k0);
      v8bf h1 = *(const v8bf*)(bh + k0 + 8);
      v8bf l0 = *(const v8bf*)(bl + k0);
      v8bf l1 = *(const v8bf*)(bl + k0 + 8);
      Ahi[mt][kt] = __builtin_shufflevector(h0, h1, 0,1,2,3,4,5,6,7,8,9,10,11,12,13,14,15);
      Alo[mt][kt] = __builtin_shufflevector(l0, l1, 0,1,2,3,4,5,6,7,8,9,10,11,12,13,14,15);
    }
  }

  const int ot0 = blockIdx.x * 16 + wave * 4;
#pragma unroll
  for (int i = 0; i < 4; ++i) {
    const int ot = ot0 + i;
    if (ot >= OTILES) continue;
    const int obase = ot * 16;
    const float* wcol = w2 + (size_t)n * HID * OUT_DIM + obase + laneN;

    v8f acc[4];
#pragma unroll
    for (int mt = 0; mt < 4; ++mt) {
      v8f z = {0.f, 0.f, 0.f, 0.f, 0.f, 0.f, 0.f, 0.f};
      acc[mt] = z;
    }

#pragma unroll
    for (int kt = 0; kt < 4; ++kt) {
      // Build B operand (32x16 tile): lane owns column obase+laneN, K packed 2/VGPR.
      v16bf Bhi, Blo;
#pragma unroll
      for (int j = 0; j < 8; ++j) {
        const int k = kt * 32 + laneH * 16 + 2 * j;
        float f0 = wcol[(size_t)k * OUT_DIM];
        float f1 = wcol[(size_t)(k + 1) * OUT_DIM];
        __bf16 bh0, bl0, bh1, bl1;
        split_bf16(f0, bh0, bl0);
        split_bf16(f1, bh1, bl1);
        Bhi[2 * j]     = bh0;
        Bhi[2 * j + 1] = bh1;
        Blo[2 * j]     = bl0;
        Blo[2 * j + 1] = bl1;
      }
      // D = A*B + C ; bf16x3: hi*hi + hi*lo + lo*hi  (f32 accumulate)
#pragma unroll
      for (int mt = 0; mt < 4; ++mt) {
        acc[mt] = __builtin_amdgcn_wmma_f32_16x16x32_bf16(
            false, Ahi[mt][kt], false, Bhi, (short)0, acc[mt], false, false);
        acc[mt] = __builtin_amdgcn_wmma_f32_16x16x32_bf16(
            false, Ahi[mt][kt], false, Blo, (short)0, acc[mt], false, false);
        acc[mt] = __builtin_amdgcn_wmma_f32_16x16x32_bf16(
            false, Alo[mt][kt], false, Bhi, (short)0, acc[mt], false, false);
      }
    }

    const float bias = b2[n * OUT_DIM + obase + laneN];
    float* orow = out + (size_t)n * BATCH * OUT_DIM + obase + laneN;
#pragma unroll
    for (int mt = 0; mt < 4; ++mt) {
#pragma unroll
      for (int j = 0; j < 8; ++j) {
        const int m = mt * 16 + j + laneH * 8;   // C layout: VGPR j -> M=j (+8 upper half)
        orow[(size_t)m * OUT_DIM] = acc[mt][j] + bias;
      }
    }
  }
}

extern "C" void kernel_launch(void* const* d_in, const int* in_sizes, int n_in,
                              void* d_out, int out_size, void* d_ws, size_t ws_size,
                              hipStream_t stream) {
  const float* x  = (const float*)d_in[0];
  const float* w1 = (const float*)d_in[1];
  const float* b1 = (const float*)d_in[2];
  const float* w2 = (const float*)d_in[3];
  const float* b2 = (const float*)d_in[4];
  float* out = (float*)d_out;

  __bf16* h_hi = (__bf16*)d_ws;                       // 24*64*128 bf16 = 384 KB
  __bf16* h_lo = h_hi + (size_t)N_DEC * BATCH * HID;  // + 384 KB

  dim3 g1(BATCH, N_DEC);
  gen_hidden_kernel<<<g1, 128, 0, stream>>>(x, w1, b1, h_hi, h_lo);

  dim3 g2((OTILES + 15) / 16, N_DEC);
  gen_out_kernel<<<g2, 128, 0, stream>>>(h_hi, h_lo, w2, b2, out);
}
